// HyperMod_89988154785842
// MI455X (gfx1250) — matrix operand order repaired and checked by reading, not compiled, not run on hardware
//
#include <hip/hip_runtime.h>

typedef __attribute__((ext_vector_type(16))) __bf16 v16bf;
typedef __attribute__((ext_vector_type(8)))  float  v8f;

#define NHD 512
#define NV_ 50000
#define NE_ 10000
#define NI_ 300000

// ---------------------------------------------------------------------------
// Prep: convert W (512x512 f32, row-major) into hi/lo bf16 pairs, pre-swizzled
// into the CDNA5 16-bit B-matrix (32x16 KxN) VGPR fragment layout:
//   fragment (kc, ntile): element [lane][j] = W[kc*32 + 8*(lane>>4) + (j&7) + 16*(j>>3)][ntile*16 + (lane&15)]
// stored contiguously so a lane reads its v16bf with one 32B load.
// ---------------------------------------------------------------------------
__global__ __launch_bounds__(256) void prep_w_kernel(
    const float* __restrict__ W1, const float* __restrict__ W2,
    __bf16* __restrict__ Whi1, __bf16* __restrict__ Wlo1,
    __bf16* __restrict__ Whi2, __bf16* __restrict__ Wlo2)
{
  int t = blockIdx.x * 256 + threadIdx.x;           // 0 .. 2*262144-1
  const float* W = W1;
  __bf16* Whi = Whi1; __bf16* Wlo = Wlo1;
  if (t >= NHD * NHD) { t -= NHD * NHD; W = W2; Whi = Whi2; Wlo = Wlo2; }
  int frag   = t >> 9;          // kc*32 + ntile
  int within = t & 511;
  int lane = within >> 4;
  int j    = within & 15;
  int kc = frag >> 5, ntile = frag & 31;
  int k   = kc * 32 + ((lane >> 4) << 3) + (j & 7) + ((j >> 3) << 4);
  int col = ntile * 16 + (lane & 15);
  float w = W[k * NHD + col];
  __bf16 hi = (__bf16)w;
  Whi[t] = hi;
  Wlo[t] = (__bf16)(w - (float)hi);
}

// ---------------------------------------------------------------------------
// Fused GEMM:  Out = relu(A @ W + bias) * rowW     (A: Mx512 f32, W: 512x512)
// bf16x3 split precision: acc += Ahi*Bhi + Alo*Bhi + Ahi*Blo  (f32 accum)
// Block = 8 waves; wave w owns m-tile (blockIdx.x*8 + w), 8 n-tiles (128 cols).
// B fragments double-buffered in LDS via GLOBAL_LOAD_ASYNC_TO_LDS_B128:
// chunk kc+1's 16 KB DMA runs under chunk kc's 24 WMMAs (ASYNCcnt-tracked).
// ---------------------------------------------------------------------------
__global__ __launch_bounds__(256) void gemm_relu_scale_kernel(
    const float* __restrict__ A, const __bf16* __restrict__ Whi,
    const __bf16* __restrict__ Wlo, const float* __restrict__ bias,
    const float* __restrict__ rowW, float* __restrict__ Out, int Mtiles)
{
  __shared__ __align__(32) __bf16 sBhi[2][8 * 512];
  __shared__ __align__(32) __bf16 sBlo[2][8 * 512];

  const int tid  = threadIdx.x;
  const int wave = __builtin_amdgcn_readfirstlane(tid >> 5);  // wave-uniform (SGPR)
  const int lane = tid & 31;
  const int h    = lane >> 4;      // lane half (K-half select)
  const int lm   = lane & 15;
  const int mtile  = blockIdx.x * 8 + wave;
  const bool active = (mtile < Mtiles);
  const int nbase  = blockIdx.y * 128;          // 8 n-tiles of 16 columns

  v8f acc[8];
#pragma unroll
  for (int tN = 0; tN < 8; ++tN)
#pragma unroll
    for (int q = 0; q < 8; ++q) acc[tN][q] = 0.0f;

  const size_t rowA = (size_t)(mtile * 16 + lm) * NHD;

  // Per-thread staging addresses (32B of hi + 32B of lo per thread per chunk).
  const int e0 = tid * 16;
  const unsigned ldsh[2] = { (unsigned)(size_t)&sBhi[0][e0],
                             (unsigned)(size_t)&sBhi[1][e0] };
  const unsigned ldsl[2] = { (unsigned)(size_t)&sBlo[0][e0],
                             (unsigned)(size_t)&sBlo[1][e0] };

  // Issue one K-chunk's B fragments (hi+lo, 16 KB) as async DMA into LDS.
  // INST_OFFSET is added to both the global and LDS addresses, so one base
  // pair covers two 16B transfers.
  auto issue = [&](int kc, int buf) {
    const size_t fbase = (size_t)(kc * 32 + blockIdx.y * 8) * 512; // 4096 bf16
    const __bf16* ghi = &Whi[fbase + e0];
    const __bf16* glo = &Wlo[fbase + e0];
    asm volatile(
        "global_load_async_to_lds_b128 %0, %2, off\n\t"
        "global_load_async_to_lds_b128 %0, %2, off offset:16\n\t"
        "global_load_async_to_lds_b128 %1, %3, off\n\t"
        "global_load_async_to_lds_b128 %1, %3, off offset:16"
        :: "v"(ldsh[buf]), "v"(ldsl[buf]), "v"(ghi), "v"(glo)
        : "memory");
  };

  issue(0, 0);

  for (int kc = 0; kc < 16; ++kc) {
    const int buf = kc & 1;
    // Wait for this chunk's DMA; barrier also guarantees every wave is done
    // reading the *other* buffer before we overwrite it below.
    asm volatile("s_wait_asynccnt 0" ::: "memory");
    __syncthreads();
    if (kc + 1 < 16) issue(kc + 1, (kc + 1) & 1);  // overlaps with compute

    if (active) {
      // ---- A fragment: 16-bit A 16x32 layout: K = (j&7) + 8h + 16*(j>>3) ----
      const float* ar = A + rowA + kc * 32 + h * 8;
      union { float4 v[4]; float f[16]; } au;
      au.v[0] = *(const float4*)(ar + 0);
      au.v[1] = *(const float4*)(ar + 4);
      au.v[2] = *(const float4*)(ar + 16);
      au.v[3] = *(const float4*)(ar + 20);
      v16bf ahi, alo;
#pragma unroll
      for (int j = 0; j < 16; ++j) {
        __bf16 hi = (__bf16)au.f[j];
        ahi[j] = hi;
        alo[j] = (__bf16)(au.f[j] - (float)hi);
      }
      // ---- 8 n-tiles x 3 WMMAs (bf16x3) ----
#pragma unroll
      for (int tN = 0; tN < 8; ++tN) {
        v16bf bhi = *(const v16bf*)&sBhi[buf][tN * 512 + lane * 16];
        v16bf blo = *(const v16bf*)&sBlo[buf][tN * 512 + lane * 16];
        acc[tN] = __builtin_amdgcn_wmma_f32_16x16x32_bf16(
            false, ahi, false, bhi, (short)0, acc[tN], false, false);
        acc[tN] = __builtin_amdgcn_wmma_f32_16x16x32_bf16(
            false, alo, false, bhi, (short)0, acc[tN], false, false);
        acc[tN] = __builtin_amdgcn_wmma_f32_16x16x32_bf16(
            false, ahi, false, blo, (short)0, acc[tN], false, false);
      }
    }
  }

  if (active) {
    // C/D layout: VGPR r -> row = mtile*16 + 8h + r, col = nbase + tN*16 + lm
    float rw[8];
#pragma unroll
    for (int r = 0; r < 8; ++r) rw[r] = rowW[mtile * 16 + h * 8 + r];
#pragma unroll
    for (int tN = 0; tN < 8; ++tN) {
      const int col = nbase + tN * 16 + lm;
      const float b = bias[col];
#pragma unroll
      for (int r = 0; r < 8; ++r) {
        float vv = acc[tN][r] + b;
        vv = vv > 0.0f ? vv : 0.0f;
        Out[(size_t)(mtile * 16 + h * 8 + r) * NHD + col] = vv * rw[r];
      }
    }
  }
}

// ---------------------------------------------------------------------------
// Scatter: dst[sidx[i]] += src[gidx[i]] * regw[i]   (one block per incidence)
// ---------------------------------------------------------------------------
__global__ __launch_bounds__(128) void scatter_add_kernel(
    const float* __restrict__ src, const int* __restrict__ gidx,
    const int* __restrict__ sidx, const float* __restrict__ regw,
    float* __restrict__ dst)
{
  const int i = blockIdx.x;
  const int g = gidx[i];
  const int s = sidx[i];
  const float w = regw[i];
  const int c = threadIdx.x * 4;
  float4 x = *(const float4*)&src[(size_t)g * NHD + c];
  float* d = &dst[(size_t)s * NHD + c];
  unsafeAtomicAdd(d + 0, x.x * w);
  unsafeAtomicAdd(d + 1, x.y * w);
  unsafeAtomicAdd(d + 2, x.z * w);
  unsafeAtomicAdd(d + 3, x.w * w);
}

// out = in * w[row]   (float4 per thread; n divisible by 1024)
__global__ __launch_bounds__(256) void row_scale_init_kernel(
    const float* __restrict__ in, const float* __restrict__ w,
    float* __restrict__ out)
{
  const size_t base = ((size_t)blockIdx.x * 256 + threadIdx.x) * 4;
  const float s = w[base >> 9];
  float4 t = *(const float4*)&in[base];
  t.x *= s; t.y *= s; t.z *= s; t.w *= s;
  *(float4*)&out[base] = t;
}

// out = in
__global__ __launch_bounds__(256) void copy_kernel(
    const float* __restrict__ in, float* __restrict__ out)
{
  const size_t base = ((size_t)blockIdx.x * 256 + threadIdx.x) * 4;
  *(float4*)&out[base] = *(const float4*)&in[base];
}

// x[row][*] /= s[row]
__global__ __launch_bounds__(256) void row_div_kernel(
    float* __restrict__ x, const float* __restrict__ s)
{
  const size_t base = ((size_t)blockIdx.x * 256 + threadIdx.x) * 4;
  const float inv = 1.0f / s[base >> 9];
  float4 t = *(const float4*)&x[base];
  t.x *= inv; t.y *= inv; t.z *= inv; t.w *= inv;
  *(float4*)&x[base] = t;
}

// ---------------------------------------------------------------------------
extern "C" void kernel_launch(void* const* d_in, const int* in_sizes, int n_in,
                              void* d_out, int out_size, void* d_ws, size_t ws_size,
                              hipStream_t stream) {
  (void)in_sizes; (void)n_in; (void)out_size; (void)ws_size;

  const float* v          = (const float*)d_in[0];
  const float* e          = (const float*)d_in[1];
  const int*   player_idx = (const int*)d_in[2];
  const int*   game_idx   = (const int*)d_in[3];
  /* idx = d_in[4] is arange(NI): identity, unused */
  const float* W_v2e      = (const float*)d_in[5];
  const float* W_e2v      = (const float*)d_in[6];
  const float* b_v        = (const float*)d_in[7];
  const float* b_e        = (const float*)d_in[8];
  const float* v_weight   = (const float*)d_in[9];
  const float* e_weight   = (const float*)d_in[10];
  const float* v_reg_w    = (const float*)d_in[11];
  const float* e_reg_w    = (const float*)d_in[12];
  const float* e_reg_sum  = (const float*)d_in[13];
  const float* v_reg_sum  = (const float*)d_in[14];

  float* v_out = (float*)d_out;                          // [NV, 512]
  float* e_out = (float*)d_out + (size_t)NV_ * NHD;      // [NE, 512]

  char* ws = (char*)d_ws;
  float*  ve_s = (float*)ws;                                         // 102.4 MB
  float*  ev_s = (float*)(ws + (size_t)NV_ * NHD * 4);               //  20.5 MB
  char*   wsw  = ws + (size_t)(NV_ + NE_) * NHD * 4;
  __bf16* Whi1 = (__bf16*)(wsw);
  __bf16* Wlo1 = (__bf16*)(wsw + 1 * NHD * NHD * 2);
  __bf16* Whi2 = (__bf16*)(wsw + 2 * NHD * NHD * 2);
  __bf16* Wlo2 = (__bf16*)(wsw + 3 * NHD * NHD * 2);

  // 0) swizzle+split both weight matrices into bf16 hi/lo fragment layout
  prep_w_kernel<<<(2 * NHD * NHD) / 256, 256, 0, stream>>>(
      W_v2e, W_e2v, Whi1, Wlo1, Whi2, Wlo2);

  // 1) init outputs:  v_out = v * v_weight ;  e_out = e
  row_scale_init_kernel<<<(NV_ * NHD / 4) / 256, 256, 0, stream>>>(v, v_weight, v_out);
  copy_kernel<<<(NE_ * NHD / 4) / 256, 256, 0, stream>>>(e, e_out);

  // 2) ve_s = relu(v @ W_v2e + b_v) * v_weight      (3125 m-tiles)
  gemm_relu_scale_kernel<<<dim3((3125 + 7) / 8, 4), 256, 0, stream>>>(
      v, Whi1, Wlo1, b_v, v_weight, ve_s, 3125);

  // 3) e_out += scatter(ve_s[player_idx] * v_reg_w) by game_idx; then /= e_reg_sum
  scatter_add_kernel<<<NI_, 128, 0, stream>>>(ve_s, player_idx, game_idx, v_reg_w, e_out);
  row_div_kernel<<<(NE_ * NHD / 4) / 256, 256, 0, stream>>>(e_out, e_reg_sum);

  // 4) ev_s = relu(e_out @ W_e2v + b_e) * e_weight   (625 m-tiles)
  gemm_relu_scale_kernel<<<dim3((625 + 7) / 8, 4), 256, 0, stream>>>(
      e_out, Whi2, Wlo2, b_e, e_weight, ev_s, 625);

  // 5) v_out += scatter(ev_s[game_idx] * e_reg_w) by player_idx; then /= v_reg_sum
  scatter_add_kernel<<<NI_, 128, 0, stream>>>(ev_s, game_idx, player_idx, e_reg_w, v_out);
  row_div_kernel<<<(NV_ * NHD / 4) / 256, 256, 0, stream>>>(v_out, v_reg_sum);
}
